// LinearAttention_90417651515419
// MI455X (gfx1250) — matrix operand order repaired and checked by reading, not compiled, not run on hardware
//
#include <hip/hip_runtime.h>
#include <hip/hip_bf16.h>

// cosFormer linear attention for MI455X (gfx1250), wave32 + WMMA bf16.
// B=4, H=8, S=8192, D=64. Memory-bound (AI ~33 flop/B, ~11us at 23.3TB/s):
// q/k/v each read exactly once; 1MB of KV intermediates live in d_ws (L2).
// Phase 1 is software-pipelined: next chunk's global loads are issued into
// registers before the WMMAs on the current LDS buffer, converted/stored to
// the ping-pong buffer afterwards (one barrier per chunk).

typedef __attribute__((ext_vector_type(16))) __bf16 v16bf;
typedef __attribute__((ext_vector_type(8)))  __bf16 v8bf;
typedef __attribute__((ext_vector_type(8)))  float  v8f;

#define SEQ   8192
#define DIM   64
#define BHTOT 32
#define SPLIT 16          // S-splits in phase 1
#define CHUNK 64          // timesteps staged per iteration
#define P1THREADS 512
// (pi/2)/8192
#define THETA_SCALE 1.9174759848570515e-4f

static __device__ inline v16bf combine8(v8bf lo, v8bf hi) {
  v16bf r;
#pragma unroll
  for (int i = 0; i < 8; ++i) { r[i] = lo[i]; r[i + 8] = hi[i]; }
  return r;
}

static __device__ inline v8f wmma_bf16(v16bf a, v16bf b, v8f c) {
  // D = A(16x32 bf16) * B(32x16 bf16) + C(16x16 f32)
  return __builtin_amdgcn_wmma_f32_16x16x32_bf16(
      /*neg_a=*/false, a, /*neg_b=*/false, b,
      /*c_mod=*/(short)0, c, /*reuse_a=*/false, /*reuse_b=*/false);
}

// ---------------------------------------------------------------------------
// Phase 0: zero the KV accumulation workspace (32 * 2 * 64*64 floats = 1MB).
// ---------------------------------------------------------------------------
__global__ void __launch_bounds__(256)
la_zero_ws(float4* __restrict__ p) {
  p[(size_t)blockIdx.x * blockDim.x + threadIdx.x] =
      make_float4(0.f, 0.f, 0.f, 0.f);
}

// ---------------------------------------------------------------------------
// Phase 1 helpers: 64-step chunk = 64x64 f32 of k and v.
// Staging layout in LDS is TRANSPOSED ([d][s] / [e][s]) bf16, rows padded
// 64 -> 72 (144B stride = 36 banks, coprime walk) so every WMMA fragment
// gather is a contiguous 16B ds_load_b128.
// ---------------------------------------------------------------------------
static __device__ inline void p1_load(const float* __restrict__ kb,
                                      const float* __restrict__ vb,
                                      int cb, int tid,
                                      float4 kq[2], float4 vq[2]) {
#pragma unroll
  for (int p = 0; p < 2; ++p) {
    const int i = tid + p * P1THREADS;          // 1024 float4s per array
    const size_t off = ((size_t)(cb + (i >> 4))) * DIM + (size_t)(i & 15) * 4;
    kq[p] = *(const float4*)&kb[off];           // coalesced global_load_b128
    vq[p] = *(const float4*)&vb[off];
  }
}

static __device__ inline void p1_store(__bf16 (*kc)[72], __bf16 (*ks)[72],
                                       __bf16 (*vt)[72],
                                       int cb, int tid,
                                       const float4 kq[2], const float4 vq[2]) {
#pragma unroll
  for (int p = 0; p < 2; ++p) {
    const int i  = tid + p * P1THREADS;
    const int sl = i >> 4;                      // timestep within chunk
    const int d0 = (i & 15) * 4;                // feature base
    float sn, cs;
    __sincosf(THETA_SCALE * (float)(cb + sl), &sn, &cs);
    const float kk[4] = {kq[p].x, kq[p].y, kq[p].z, kq[p].w};
    const float vv[4] = {vq[p].x, vq[p].y, vq[p].z, vq[p].w};
#pragma unroll
    for (int j = 0; j < 4; ++j) {
      const float kr = fmaxf(kk[j], 0.0f);
      kc[d0 + j][sl] = (__bf16)(kr * cs);
      ks[d0 + j][sl] = (__bf16)(kr * sn);
      vt[d0 + j][sl] = (__bf16)vv[j];
    }
  }
}

static __device__ inline void p1_compute(const __bf16 (*kc)[72],
                                         const __bf16 (*ks)[72],
                                         const __bf16 (*vt)[72],
                                         int dRow, int eCol, int half,
                                         v8f& accC, v8f& accS) {
#pragma unroll
  for (int kcn = 0; kcn < 2; ++kcn) {           // two K=32 sub-chunks
    const int aoff = kcn * 32 + half * 8;
    const v16bf aC = combine8(*(const v8bf*)&kc[dRow][aoff],
                              *(const v8bf*)&kc[dRow][aoff + 16]);
    const v16bf aS = combine8(*(const v8bf*)&ks[dRow][aoff],
                              *(const v8bf*)&ks[dRow][aoff + 16]);
    const int boff = kcn * 32 + half * 8;
    const v16bf bV = combine8(*(const v8bf*)&vt[eCol][boff],
                              *(const v8bf*)&vt[eCol][boff + 16]);
    accC = wmma_bf16(aC, bV, accC);
    accS = wmma_bf16(aS, bV, accS);
  }
}

// ---------------------------------------------------------------------------
// Phase 1: kv_cos[bh] = (relu(k)*cos)^T V ; kv_sin likewise. [64 x 64] each.
// grid = (BH=32, SPLIT=16), block = 512 (16 waves). Wave w owns output tile
// (dTile = w>>2, eTile = w&3); 8 pipelined 64-step chunks, then f32 atomics.
// ---------------------------------------------------------------------------
__global__ void __launch_bounds__(P1THREADS)
la_kv_kernel(const float* __restrict__ k, const float* __restrict__ v,
             float* __restrict__ kvws) {
  __shared__ __align__(16) __bf16 kcT[2][64][72];
  __shared__ __align__(16) __bf16 ksT[2][64][72];
  __shared__ __align__(16) __bf16 vT [2][64][72];

  const int bh   = blockIdx.x;
  const int tid  = threadIdx.x;
  const int w    = tid >> 5;
  const int lane = tid & 31;
  const int m    = lane & 15;
  const int half = lane >> 4;
  const int dRow = (w >> 2) * 16 + m;           // A-fragment row (d)
  const int eCol = (w & 3) * 16 + m;            // B-fragment col (e)

  const size_t base = (size_t)bh * SEQ * DIM;
  const float* kb = k + base;
  const float* vb = v + base;

  v8f accC = {};
  v8f accS = {};
  float4 kq[2], vq[2];

  const int slice0 = blockIdx.y * (SEQ / SPLIT);    // 512 steps per block
  constexpr int NCH = (SEQ / SPLIT) / CHUNK;        // 8 chunks

  // Prologue: stage chunk 0 into buffer 0.
  p1_load(kb, vb, slice0, tid, kq, vq);
  p1_store(kcT[0], ksT[0], vT[0], slice0, tid, kq, vq);
  __syncthreads();

  for (int it = 0; it < NCH; ++it) {
    const int cur = it & 1;
    const int nxt = cur ^ 1;
    if (it + 1 < NCH)                                // issue next loads early
      p1_load(kb, vb, slice0 + (it + 1) * CHUNK, tid, kq, vq);
    p1_compute(kcT[cur], ksT[cur], vT[cur], dRow, eCol, half, accC, accS);
    if (it + 1 < NCH)                                // convert + stage next
      p1_store(kcT[nxt], ksT[nxt], vT[nxt],
               slice0 + (it + 1) * CHUNK, tid, kq, vq);
    __syncthreads();
  }

  // C/D layout: VGPR r -> row (r + half*8), col = m. Accumulate partials.
  float* kvc = kvws + (size_t)bh * 2 * (DIM * DIM);
  float* kvs = kvc + DIM * DIM;
#pragma unroll
  for (int r = 0; r < 8; ++r) {
    const int dG = (dRow - m) + half * 8 + r;        // dTile*16 + half*8 + r
    unsafeAtomicAdd(&kvc[dG * DIM + eCol], accC[r]);
    unsafeAtomicAdd(&kvs[dG * DIM + eCol], accS[r]);
  }
}

// ---------------------------------------------------------------------------
// Phase 2: out = (relu(q)*cos) @ kv_cos + (relu(q)*sin) @ kv_sin.
// grid = (S/128 = 64, BH = 32), block = 256 (8 waves); wave w owns 16 rows.
// KV staged once per block into LDS transposed to [e][d] bf16 (rows 64->72).
// ---------------------------------------------------------------------------
__global__ void __launch_bounds__(256)
la_out_kernel(const float* __restrict__ q, const float* __restrict__ kvws,
              float* __restrict__ out) {
  __shared__ __align__(16) __bf16 kvcT[64][72];
  __shared__ __align__(16) __bf16 kvsT[64][72];

  const int bh     = blockIdx.y;
  const int sBlock = blockIdx.x * 128;
  const int tid    = threadIdx.x;

  const float* kvc = kvws + (size_t)bh * 2 * (DIM * DIM);
  const float* kvs = kvc + DIM * DIM;
  for (int i = tid; i < (DIM * DIM) / 4; i += 256) { // 1024 float4s
    const int d  = i >> 4;
    const int e0 = (i & 15) * 4;
    const float4 c4 = *(const float4*)&kvc[d * DIM + e0];
    const float4 s4 = *(const float4*)&kvs[d * DIM + e0];
    kvcT[e0 + 0][d] = (__bf16)c4.x; kvcT[e0 + 1][d] = (__bf16)c4.y;
    kvcT[e0 + 2][d] = (__bf16)c4.z; kvcT[e0 + 3][d] = (__bf16)c4.w;
    kvsT[e0 + 0][d] = (__bf16)s4.x; kvsT[e0 + 1][d] = (__bf16)s4.y;
    kvsT[e0 + 2][d] = (__bf16)s4.z; kvsT[e0 + 3][d] = (__bf16)s4.w;
  }
  __syncthreads();

  const int w    = tid >> 5;
  const int lane = tid & 31;
  const int m    = lane & 15;
  const int half = lane >> 4;

  const int sRow = sBlock + w * 16 + m;
  float sn, cs;
  __sincosf(THETA_SCALE * (float)sRow, &sn, &cs);

  // Build A fragments for K-chunks kc=0,1 (d = kc*32 .. +31).
  const float* qrow = q + ((size_t)bh * SEQ + sRow) * DIM;
  v16bf aC[2], aS[2];
#pragma unroll
  for (int kc = 0; kc < 2; ++kc) {
    const int d0 = half * 8 + kc * 32;
    const float4 f0 = *(const float4*)&qrow[d0];
    const float4 f1 = *(const float4*)&qrow[d0 + 4];
    const float4 f2 = *(const float4*)&qrow[d0 + 16];
    const float4 f3 = *(const float4*)&qrow[d0 + 20];
    const float vals[16] = {f0.x, f0.y, f0.z, f0.w, f1.x, f1.y, f1.z, f1.w,
                            f2.x, f2.y, f2.z, f2.w, f3.x, f3.y, f3.z, f3.w};
#pragma unroll
    for (int e = 0; e < 16; ++e) {
      const float r = fmaxf(vals[e], 0.0f);
      aC[kc][e] = (__bf16)(r * cs);
      aS[kc][e] = (__bf16)(r * sn);
    }
  }

  v8f acc[4] = {};
#pragma unroll
  for (int et = 0; et < 4; ++et) {
    const int eCol = et * 16 + m;
#pragma unroll
    for (int kc = 0; kc < 2; ++kc) {
      const int off = kc * 32 + half * 8;
      const v16bf bC = combine8(*(const v8bf*)&kvcT[eCol][off],
                                *(const v8bf*)&kvcT[eCol][off + 16]);
      acc[et] = wmma_bf16(aC[kc], bC, acc[et]);
      const v16bf bS = combine8(*(const v8bf*)&kvsT[eCol][off],
                                *(const v8bf*)&kvsT[eCol][off + 16]);
      acc[et] = wmma_bf16(aS[kc], bS, acc[et]);
    }
  }

  // Store: row = sBlock + w*16 + half*8 + r; cols et*16 + m (64B runs/half).
  float* orow = out + ((size_t)bh * SEQ + sBlock + w * 16 + half * 8) * DIM;
#pragma unroll
  for (int r = 0; r < 8; ++r) {
#pragma unroll
    for (int et = 0; et < 4; ++et) {
      orow[(size_t)r * DIM + et * 16 + m] = acc[et][r];
    }
  }
}

// ---------------------------------------------------------------------------
extern "C" void kernel_launch(void* const* d_in, const int* in_sizes, int n_in,
                              void* d_out, int out_size, void* d_ws, size_t ws_size,
                              hipStream_t stream) {
  (void)in_sizes; (void)n_in; (void)out_size; (void)ws_size;
  const float* q = (const float*)d_in[0];
  const float* k = (const float*)d_in[1];
  const float* v = (const float*)d_in[2];
  float* out  = (float*)d_out;
  float* kvws = (float*)d_ws;   // needs 32*2*4096*4 = 1 MiB

  // Phase 0: zero KV workspace (262144 floats = 65536 float4).
  la_zero_ws<<<dim3(256), dim3(256), 0, stream>>>((float4*)d_ws);
  // Phase 1: KV = K_w^T V (cos & sin), split-S, pipelined, f32 atomics.
  la_kv_kernel<<<dim3(BHTOT, SPLIT), dim3(P1THREADS), 0, stream>>>(k, v, kvws);
  // Phase 2: out = Q_w @ KV (cos + sin).
  la_out_kernel<<<dim3(SEQ / 128, BHTOT), dim3(256), 0, stream>>>(q, kvws, out);
}
